// MultiHeadCrossAttention_57638461112647
// MI455X (gfx1250) — compile-verified, matrix operand order checked by reading, not run on hardware
//
#include <hip/hip_runtime.h>

#define DEV __device__ __forceinline__

typedef __attribute__((ext_vector_type(16))) __bf16 bfvec16;
typedef __attribute__((ext_vector_type(8)))  float  fvec8;

static constexpr int BB = 2, CS = 48, CY = 96;
static constexpr int NN = 16 * 16 * 16;          // 4096 voxels
static constexpr int QS = 4;                     // q-range split for column stats
static constexpr int KS = 4;                     // k-range split for apply pass
static constexpr float BN_EPS   = 1e-5f;
static constexpr float ATT_SCALE = 0.14433756729740643f;   // 48^-0.5

// ---------- small helpers ----------
DEV unsigned short f2bf(float f) {                 // fp32 -> bf16 bits, RNE
  unsigned int u = __builtin_bit_cast(unsigned int, f);
  u += 0x7fffu + ((u >> 16) & 1u);
  return (unsigned short)(u >> 16);
}
DEV float bf2f(unsigned short s) {
  unsigned int u = ((unsigned int)s) << 16;
  return __builtin_bit_cast(float, u);
}
DEV __bf16 us2bf(unsigned short s) { return __builtin_bit_cast(__bf16, s); }

DEV fvec8 wmma_bf16(bfvec16 a, bfvec16 b, fvec8 c) {
  // D(16x16,f32) = A(16x32,bf16) * B(32x16,bf16) + C
  return __builtin_amdgcn_wmma_f32_16x16x32_bf16(false, a, false, b,
                                                 (short)0, c, false, false);
}

// K index within a 16-deep bf16 A/B operand vector, per ISA 7.12.2 layout.
// lane group g = lane>>4; slot i -> vgpr v=i>>1, half h=i&1
DEV int wmma_kidx(int i, int g) {
  int v = i >> 1, h = i & 1;
  return ((v & 4) ? 16 : 0) + ((v & 3) << 1) + h + (g << 3);
}

// analytic positional encoding value for channel ch at voxel n
DEV float pe_val(int ch, int n, int cpe) {
  int grp = ch / cpe, j = ch % cpe, hp = cpe >> 1;
  int pos = (grp == 0) ? (n >> 8) : (grp == 1) ? ((n >> 4) & 15) : (n & 15);
  int jj = (j < hp) ? j : (j - hp);
  float f = __powf(10000.0f, -2.0f * (float)jj / (float)cpe);
  float ang = (float)pos * f;
  return (j < hp) ? __sinf(ang) : __cosf(ang);
}

// ---------- stage 1: add positional encodings ----------
__global__ void k_pe_add(const float* __restrict__ Y, const float* __restrict__ S,
                         float* __restrict__ Yy, float* __restrict__ Ss) {
  int idx = blockIdx.x * 256 + threadIdx.x;
  const int totS = BB * CS * NN, totY = BB * CY * NN;
  if (idx < totS) {
    int n = idx % NN, c = (idx / NN) % CS;
    Ss[idx] = S[idx] + pe_val(c, n, 16);
  } else if (idx < totS + totY) {
    int j = idx - totS;
    int n = j % NN, c = (j / NN) % CY;
    Yy[j] = Y[j] + pe_val(c, n, 32);
  }
}

// ---------- conv1x1 + BN + ReLU, (b,Cin,N) -> (b,N,CS) ----------
__global__ void k_proj(const float* __restrict__ in, const float* __restrict__ w,
                       const float* __restrict__ bias, const float* __restrict__ gam,
                       const float* __restrict__ bet, float* __restrict__ out, int Cin) {
  int idx = blockIdx.x * 256 + threadIdx.x;          // ((b*CS+o)*NN + n)
  if (idx >= BB * CS * NN) return;
  int n = idx % NN, o = (idx / NN) % CS, b = idx / (CS * NN);
  const float* ip = in + (size_t)b * Cin * NN + n;
  const float* wp = w + o * Cin;
  float acc = bias[o];
  for (int c = 0; c < Cin; ++c) acc += wp[c] * ip[(size_t)c * NN];
  float sc = gam[o] * rsqrtf(1.0f + BN_EPS);
  out[((size_t)b * NN + n) * CS + o] = fmaxf(sc * acc + bet[o], 0.0f);
}

// ---------- Q/K/V projection, bf16 output (b,N,CS) ----------
__global__ void k_qkv(const float* __restrict__ Y1, const float* __restrict__ S1,
                      const float* __restrict__ Wq, const float* __restrict__ Wk,
                      const float* __restrict__ Wv,
                      unsigned short* __restrict__ Qb, unsigned short* __restrict__ Kb,
                      unsigned short* __restrict__ Vb) {
  int idx = blockIdx.x * 256 + threadIdx.x;          // (b*NN+n)*CS + e
  if (idx >= BB * NN * CS) return;
  int e = idx % CS;  size_t bn = (size_t)(idx / CS);
  const float* yr = Y1 + bn * CS;
  const float* sr = S1 + bn * CS;
  float q = 0.f, k = 0.f, v = 0.f;
  for (int d = 0; d < CS; ++d) {
    q += yr[d] * Wq[d * CS + e];
    k += yr[d] * Wk[d * CS + e];
    v += sr[d] * Wv[d * CS + e];
  }
  Qb[idx] = f2bf(q); Kb[idx] = f2bf(k); Vb[idx] = f2bf(v);
}

// ---------- 3x3x3 conv, SAME, 96->96 ----------
__global__ void k_conv3(const float* __restrict__ Yy, const float* __restrict__ w3,
                        const float* __restrict__ b3, float* __restrict__ Y3) {
  int idx = blockIdx.x * 256 + threadIdx.x;
  if (idx >= BB * CY * NN) return;
  int n = idx % NN, o = (idx / NN) % CY, b = idx / (CY * NN);
  int d = n >> 8, h = (n >> 4) & 15, w = n & 15;
  float acc = b3[o];
  for (int ci = 0; ci < CY; ++ci) {
    const float* ip = Yy + ((size_t)(b * CY + ci)) * NN;
    const float* wp = w3 + ((size_t)(o * CY + ci)) * 27;
#pragma unroll
    for (int kd = 0; kd < 3; ++kd) {
      int dz = d + kd - 1; if ((unsigned)dz >= 16u) continue;
#pragma unroll
      for (int kh = 0; kh < 3; ++kh) {
        int hz = h + kh - 1; if ((unsigned)hz >= 16u) continue;
#pragma unroll
        for (int kw = 0; kw < 3; ++kw) {
          int wz = w + kw - 1; if ((unsigned)wz >= 16u) continue;
          acc += wp[(kd * 3 + kh) * 3 + kw] * ip[(dz << 8) + (hz << 4) + wz];
        }
      }
    }
  }
  Y3[idx] = acc;
}

// ---------- Y2 = ReLU(BN(conv1x1(Y3))) into out channels [48,96) ----------
__global__ void k_y2(const float* __restrict__ Y3, const float* __restrict__ w,
                     const float* __restrict__ bias, const float* __restrict__ gam,
                     const float* __restrict__ bet, float* __restrict__ out) {
  int idx = blockIdx.x * 256 + threadIdx.x;          // ((b*CS+s)*NN + n)
  if (idx >= BB * CS * NN) return;
  int n = idx % NN, s = (idx / NN) % CS, b = idx / (CS * NN);
  const float* ip = Y3 + (size_t)b * CY * NN + n;
  const float* wp = w + s * CY;
  float acc = bias[s];
  for (int o = 0; o < CY; ++o) acc += wp[o] * ip[(size_t)o * NN];
  float sc = gam[s] * rsqrtf(1.0f + BN_EPS);
  out[((size_t)(b * 2 * CS) + CS + s) * NN + n] = fmaxf(sc * acc + bet[s], 0.0f);
}

// ---------- attention pass 1: per-key-column max/sum of exp (softmax over q) ----
__global__ void k_colstats(const unsigned short* __restrict__ Qb,
                           const unsigned short* __restrict__ Kb,
                           float* __restrict__ mPart, float* __restrict__ lPart) {
  const int tid = threadIdx.x, lane = tid & 31, wv = tid >> 5;
  const int kTiles = NN / 128;
  int blk = blockIdx.x;
  int qs = blk % QS, kt = (blk / QS) % kTiles, b = blk / (QS * kTiles);
  int kw0 = kt * 128 + wv * 16;                      // this wave's 16 keys
  int g = lane >> 4, nn = lane & 15;

  __shared__ unsigned short lq[32 * CS];

  // B operands: K^T tile for this wave's keys (d padded 48 -> 64)
  bfvec16 bk0, bk1;
  const unsigned short* kr = Kb + ((size_t)(b * NN + kw0 + nn)) * CS;
#pragma unroll
  for (int i = 0; i < 16; ++i) {
    int kk = wmma_kidx(i, g);
    bk0[i] = us2bf(kr[kk]);
    int d1 = kk + 32;
    bk1[i] = (d1 < CS) ? us2bf(kr[d1]) : us2bf((unsigned short)0);
  }

  float mrun = -3.0e38f, lrun = 0.0f;
  const int qA = qs * (NN / QS), qB = qA + NN / QS;
  for (int q0 = qA; q0 < qB; q0 += 32) {
    __syncthreads();
    for (int i = tid; i < 32 * CS; i += 256)
      lq[i] = Qb[((size_t)(b * NN + q0)) * CS + i];
    __syncthreads();
#pragma unroll
    for (int hf = 0; hf < 2; ++hf) {
      const unsigned short* row = lq + (hf * 16 + nn) * CS;   // A row m = lane&15
      bfvec16 a0, a1;
#pragma unroll
      for (int i = 0; i < 16; ++i) {
        int kk = wmma_kidx(i, g);
        a0[i] = us2bf(row[kk]);
        int d1 = kk + 32;
        a1[i] = (d1 < CS) ? us2bf(row[d1]) : us2bf((unsigned short)0);
      }
      fvec8 c;
#pragma unroll
      for (int r = 0; r < 8; ++r) c[r] = 0.0f;
      c = wmma_bf16(a0, bk0, c);
      c = wmma_bf16(a1, bk1, c);
      // online column (over q) max & exp-sum; column n held by lanes n, n+16
      float tm = -3.0e38f;
#pragma unroll
      for (int r = 0; r < 8; ++r) tm = fmaxf(tm, c[r] * ATT_SCALE);
      tm = fmaxf(tm, __shfl_xor(tm, 16, 32));
      float mnew = fmaxf(mrun, tm);
      float p = 0.0f;
#pragma unroll
      for (int r = 0; r < 8; ++r) p += __expf(c[r] * ATT_SCALE - mnew);
      p += __shfl_xor(p, 16, 32);
      lrun = lrun * __expf(mrun - mnew) + p;
      mrun = mnew;
    }
  }
  if (lane < 16) {
    size_t o = ((size_t)(qs * BB + b)) * NN + kw0 + lane;
    mPart[o] = mrun; lPart[o] = lrun;
  }
}

__global__ void k_combine(const float* __restrict__ mPart, const float* __restrict__ lPart,
                          float* __restrict__ mArr, float* __restrict__ rlArr) {
  int idx = blockIdx.x * 256 + threadIdx.x;
  if (idx >= BB * NN) return;
  int b = idx / NN, k = idx % NN;
  float m = -3.0e38f;
  for (int qs = 0; qs < QS; ++qs)
    m = fmaxf(m, mPart[((size_t)(qs * BB + b)) * NN + k]);
  float l = 0.0f;
  for (int qs = 0; qs < QS; ++qs) {
    size_t o = ((size_t)(qs * BB + b)) * NN + k;
    l += lPart[o] * __expf(mPart[o] - m);
  }
  mArr[idx] = m; rlArr[idx] = 1.0f / l;
}

// ---------- attention pass 2: x = (exp(S-m)/l) @ V, k-range split -------------
__global__ void k_apply(const unsigned short* __restrict__ Qb,
                        const unsigned short* __restrict__ Kb,
                        const unsigned short* __restrict__ Vb,
                        const float* __restrict__ mArr, const float* __restrict__ rlArr,
                        float* __restrict__ Xpart) {
  const int tid = threadIdx.x, lane = tid & 31, wv = tid >> 5;
  const int qTiles = NN / 128;
  int blk = blockIdx.x;
  int ks = blk % KS, qt = (blk / KS) % qTiles, b = blk / (KS * qTiles);
  int qw0 = qt * 128 + wv * 16;                      // this wave's 16 query rows
  int g = lane >> 4, nn = lane & 15;

  __shared__ unsigned short lk[32 * CS];
  __shared__ unsigned short lv[32 * CS];
  __shared__ float lm[32];
  __shared__ unsigned short lp[8 * 16 * 32];         // per-wave P tile (bf16)

  // A operands: this wave's Q rows (resident for all k iterations)
  bfvec16 aq0, aq1;
  const unsigned short* qr = Qb + ((size_t)(b * NN + qw0 + nn)) * CS;
#pragma unroll
  for (int i = 0; i < 16; ++i) {
    int kk = wmma_kidx(i, g);
    aq0[i] = us2bf(qr[kk]);
    int d1 = kk + 32;
    aq1[i] = (d1 < CS) ? us2bf(qr[d1]) : us2bf((unsigned short)0);
  }
  fvec8 x0, x1, x2;
#pragma unroll
  for (int r = 0; r < 8; ++r) { x0[r] = 0.f; x1[r] = 0.f; x2[r] = 0.f; }

  unsigned short* lpw = lp + wv * 512;
  const int kA = ks * (NN / KS), kB = kA + NN / KS;
  for (int k0 = kA; k0 < kB; k0 += 32) {
    __syncthreads();
    for (int i = tid; i < 32 * CS; i += 256) {
      lk[i] = Kb[((size_t)(b * NN + k0)) * CS + i];
      int krow = i / CS;
      float vv = bf2f(Vb[((size_t)(b * NN + k0)) * CS + i]) *
                 rlArr[(size_t)b * NN + k0 + krow];        // fold 1/l into V rows
      lv[i] = f2bf(vv);
    }
    if (tid < 32) lm[tid] = mArr[(size_t)b * NN + k0 + tid];
    __syncthreads();

    // scores -> P tile (16q x 32k) in per-wave LDS scratch
#pragma unroll
    for (int hf = 0; hf < 2; ++hf) {
      const unsigned short* krow = lk + (hf * 16 + nn) * CS;
      bfvec16 bk0, bk1;
#pragma unroll
      for (int i = 0; i < 16; ++i) {
        int kk = wmma_kidx(i, g);
        bk0[i] = us2bf(krow[kk]);
        int d1 = kk + 32;
        bk1[i] = (d1 < CS) ? us2bf(krow[d1]) : us2bf((unsigned short)0);
      }
      fvec8 c;
#pragma unroll
      for (int r = 0; r < 8; ++r) c[r] = 0.0f;
      c = wmma_bf16(aq0, bk0, c);
      c = wmma_bf16(aq1, bk1, c);
      float mc = lm[hf * 16 + nn];
#pragma unroll
      for (int r = 0; r < 8; ++r) {
        float pv = __expf(c[r] * ATT_SCALE - mc);
        lpw[(r + 8 * g) * 32 + hf * 16 + nn] = f2bf(pv);   // C layout -> [q][k]
      }
    }
#if __has_builtin(__builtin_amdgcn_wave_barrier)
    __builtin_amdgcn_wave_barrier();
#endif
    // repack P as A operand, then x += P(16x32) @ V'(32x48)
    bfvec16 ap;
#pragma unroll
    for (int i = 0; i < 16; ++i)
      ap[i] = us2bf(lpw[nn * 32 + wmma_kidx(i, g)]);
    {
      bfvec16 bv;
#pragma unroll
      for (int i = 0; i < 16; ++i) bv[i] = us2bf(lv[wmma_kidx(i, g) * CS + nn]);
      x0 = wmma_bf16(ap, bv, x0);
    }
    {
      bfvec16 bv;
#pragma unroll
      for (int i = 0; i < 16; ++i) bv[i] = us2bf(lv[wmma_kidx(i, g) * CS + 16 + nn]);
      x1 = wmma_bf16(ap, bv, x1);
    }
    {
      bfvec16 bv;
#pragma unroll
      for (int i = 0; i < 16; ++i) bv[i] = us2bf(lv[wmma_kidx(i, g) * CS + 32 + nn]);
      x2 = wmma_bf16(ap, bv, x2);
    }
  }
  size_t base = ((size_t)(ks * BB + b)) * NN;
#pragma unroll
  for (int r = 0; r < 8; ++r) {
    int m = r + 8 * g;
    size_t rowo = (base + qw0 + m) * CS;
    Xpart[rowo + nn]      = x0[r];
    Xpart[rowo + 16 + nn] = x1[r];
    Xpart[rowo + 32 + nn] = x2[r];
  }
}

__global__ void k_reducex(const float* __restrict__ Xpart, float* __restrict__ X) {
  int idx = blockIdx.x * 256 + threadIdx.x;
  if (idx >= BB * NN * CS) return;
  float s = 0.f;
  for (int ks = 0; ks < KS; ++ks) s += Xpart[(size_t)ks * (BB * NN * CS) + idx];
  X[idx] = s;
}

// ---------- Z = ReLU(BN(conv1x1(x))) * (S+pe) into out channels [0,48) -------
__global__ void k_final(const float* __restrict__ X, const float* __restrict__ Ss,
                        const float* __restrict__ w, const float* __restrict__ bias,
                        const float* __restrict__ gam, const float* __restrict__ bet,
                        float* __restrict__ out) {
  int idx = blockIdx.x * 256 + threadIdx.x;          // ((b*CS+o)*NN + n)
  if (idx >= BB * CS * NN) return;
  int n = idx % NN, o = (idx / NN) % CS, b = idx / (CS * NN);
  const float* xr = X + ((size_t)(b * NN + n)) * CS;
  const float* wp = w + o * CS;
  float acc = bias[o];
  for (int c = 0; c < CS; ++c) acc += wp[c] * xr[c];
  float sc = gam[o] * rsqrtf(1.0f + BN_EPS);
  float z = fmaxf(sc * acc + bet[o], 0.0f);
  out[((size_t)(b * 2 * CS) + o) * NN + n] = z * Ss[idx];
}

// ------------------------------- launcher -----------------------------------
extern "C" void kernel_launch(void* const* d_in, const int* in_sizes, int n_in,
                              void* d_out, int out_size, void* d_ws, size_t ws_size,
                              hipStream_t stream) {
  const float* Y    = (const float*)d_in[0];
  const float* S    = (const float*)d_in[1];
  const float* w_s  = (const float*)d_in[2];
  const float* b_s  = (const float*)d_in[3];
  const float* g_s  = (const float*)d_in[4];
  const float* be_s = (const float*)d_in[5];
  const float* w_y  = (const float*)d_in[6];
  const float* b_y  = (const float*)d_in[7];
  const float* g_y  = (const float*)d_in[8];
  const float* be_y = (const float*)d_in[9];
  const float* Wq   = (const float*)d_in[10];
  const float* Wk   = (const float*)d_in[11];
  const float* Wv   = (const float*)d_in[12];
  const float* w_o  = (const float*)d_in[13];
  const float* b_o  = (const float*)d_in[14];
  const float* g_o  = (const float*)d_in[15];
  const float* be_o = (const float*)d_in[16];
  const float* w3   = (const float*)d_in[17];
  const float* b3   = (const float*)d_in[18];
  const float* w_y2 = (const float*)d_in[19];
  const float* b_y2 = (const float*)d_in[20];
  const float* g_y2 = (const float*)d_in[21];
  const float* be_y2= (const float*)d_in[22];
  float* out = (float*)d_out;

  char* ws = (char*)d_ws; size_t off = 0;
  auto carveF = [&](size_t n) { float* p = (float*)(ws + off); off += n * 4; return p; };
  auto carveH = [&](size_t n) { unsigned short* p = (unsigned short*)(ws + off);
                                off += ((n * 2 + 3) / 4) * 4; return p; };
  float* Ss    = carveF((size_t)BB * CS * NN);
  float* Yy    = carveF((size_t)BB * CY * NN);
  float* S1    = carveF((size_t)BB * NN * CS);
  float* Y1    = carveF((size_t)BB * NN * CS);
  float* Y3    = carveF((size_t)BB * CY * NN);
  float* Xr    = carveF((size_t)BB * NN * CS);
  float* Xpart = carveF((size_t)KS * BB * NN * CS);
  float* mPart = carveF((size_t)QS * BB * NN);
  float* lPart = carveF((size_t)QS * BB * NN);
  float* mArr  = carveF((size_t)BB * NN);
  float* rlArr = carveF((size_t)BB * NN);
  unsigned short* Qb = carveH((size_t)BB * NN * CS);
  unsigned short* Kb = carveH((size_t)BB * NN * CS);
  unsigned short* Vb = carveH((size_t)BB * NN * CS);
  (void)off; (void)ws_size; (void)in_sizes; (void)n_in; (void)out_size;

  const dim3 blk(256);
  const int totPE = BB * (CS + CY) * NN;
  const int totP  = BB * CS * NN;
  const int totQ  = BB * NN * CS;
  const int totC  = BB * CY * NN;

  k_pe_add  <<<(totPE + 255) / 256, blk, 0, stream>>>(Y, S, Yy, Ss);
  k_proj    <<<(totP  + 255) / 256, blk, 0, stream>>>(Ss, w_s, b_s, g_s, be_s, S1, CS);
  k_proj    <<<(totP  + 255) / 256, blk, 0, stream>>>(Yy, w_y, b_y, g_y, be_y, Y1, CY);
  k_qkv     <<<(totQ  + 255) / 256, blk, 0, stream>>>(Y1, S1, Wq, Wk, Wv, Qb, Kb, Vb);
  k_conv3   <<<(totC  + 255) / 256, blk, 0, stream>>>(Yy, w3, b3, Y3);
  k_y2      <<<(totP  + 255) / 256, blk, 0, stream>>>(Y3, w_y2, b_y2, g_y2, be_y2, out);
  k_colstats<<<BB * (NN / 128) * QS, blk, 0, stream>>>(Qb, Kb, mPart, lPart);
  k_combine <<<(BB * NN + 255) / 256, blk, 0, stream>>>(mPart, lPart, mArr, rlArr);
  k_apply   <<<BB * (NN / 128) * KS, blk, 0, stream>>>(Qb, Kb, Vb, mArr, rlArr, Xpart);
  k_reducex <<<(totQ + 255) / 256, blk, 0, stream>>>(Xpart, Xr);
  k_final   <<<(totP + 255) / 256, blk, 0, stream>>>(Xr, Ss, w_o, b_o, g_o, be_o, out);
}
